// FlashOccBEVEncoder_16097537425507
// MI455X (gfx1250) — matrix operand order repaired and checked by reading, not compiled, not run on hardware
//
#include <hip/hip_runtime.h>
#include <hip/hip_bf16.h>

// ---------------------------------------------------------------------------
// FlashOcc BEV encoder for gfx1250 (MI455X).
// Convs = implicit GEMM, bf16 v_wmma_f32_16x16x32_bf16, f32 accum.
// Tile staging = global_load_async_to_lds_b128 (ASYNCcnt) + double-buffered
// LDS; activations kept in zero-padded NHWC so inner loops are branch-free.
// Block tile 128M x 128N; 8 waves (4M x 2N); wave tile 32x64 (2x4 WMMA).
// ---------------------------------------------------------------------------

typedef __bf16 bf16;
typedef __attribute__((ext_vector_type(16))) __bf16 v16bf;
typedef __attribute__((ext_vector_type(8)))  float  v8f;

struct U128x2 { uint4 a, b; };

#define EPS 1e-5f

__device__ __forceinline__ v16bf make_frag(uint4 lo, uint4 hi) {
  U128x2 u{lo, hi};
  return __builtin_bit_cast(v16bf, u);
}

// Async 16-byte copy global -> LDS (tracked by ASYNCcnt).
__device__ __forceinline__ void async_b128(unsigned lds_off, const void* gaddr) {
  asm volatile("global_load_async_to_lds_b128 %0, %1, off"
               :: "v"(lds_off), "v"(gaddr) : "memory");
}
// LDS byte offset of a __shared__ element (addrspace(3) offset = low 32 bits).
__device__ __forceinline__ unsigned lds_off(const void* p) {
  return (unsigned)(unsigned long long)p;
}
#define WAIT_ASYNC(N) asm volatile("s_wait_asynccnt %0" :: "n"(N) : "memory")

// ---------------------------------------------------------------------------
// Zero-fill (for padded buffers' halos)
// ---------------------------------------------------------------------------
__global__ __launch_bounds__(256) void zero_buf(uint4* __restrict__ p, int n16) {
  int i = blockIdx.x * 256 + threadIdx.x;
  if (i < n16) p[i] = uint4{0, 0, 0, 0};
}

// ---------------------------------------------------------------------------
// Prep kernels: layout transforms + fp32 -> bf16
// ---------------------------------------------------------------------------

// occ (8,32,200,200) NCHW fp32 -> padded (8,206,206,32) NHWC bf16 (pad 3)
__global__ __launch_bounds__(256) void prep_input(const float* __restrict__ src,
                                                  bf16* __restrict__ dst) {
  int idx = blockIdx.x * 256 + threadIdx.x;
  if (idx >= 8 * 32 * 200 * 200) return;
  int c = idx & 31;
  int w = (idx >> 5) % 200;
  int h = (idx / (32 * 200)) % 200;
  int n = idx / (32 * 200 * 200);
  dst[(((size_t)n * 206 + h + 3) * 206 + (w + 3)) * 32 + c] =
      (bf16)src[(((size_t)n * 32 + c) * 200 + h) * 200 + w];
}

// stem_w (256,32,7,7) OIHW -> [tap49][co256][ci32] bf16 (K contiguous)
__global__ __launch_bounds__(256) void prep_stem_w(const float* __restrict__ src,
                                                   bf16* __restrict__ dst) {
  int idx = blockIdx.x * 256 + threadIdx.x;
  if (idx >= 256 * 32 * 49) return;
  int tap = idx % 49;
  int ci  = (idx / 49) % 32;
  int co  = idx / (49 * 32);
  dst[((size_t)tap * 256 + co) * 32 + ci] = (bf16)src[((size_t)co * 32 + ci) * 49 + tap];
}

// blk_w (3,2,256,256,3,3) -> [b*2+cv][tap9][co256][ci256] bf16 (K contiguous)
__global__ __launch_bounds__(256) void prep_blk_w(const float* __restrict__ src,
                                                  bf16* __restrict__ dst) {
  int idx = blockIdx.x * 256 + threadIdx.x;
  if (idx >= 6 * 256 * 256 * 9) return;
  int kx = idx % 3;
  int ky = (idx / 3) % 3;
  int ci = (idx / 9) % 256;
  int co = (idx / (9 * 256)) % 256;
  int b2 = idx / (9 * 256 * 256);
  int tap = ky * 3 + kx;
  dst[(((size_t)b2 * 9 + tap) * 256 + co) * 256 + ci] =
      (bf16)src[((((size_t)b2 * 256 + co) * 256 + ci) * 3 + ky) * 3 + kx];
}

// dn_w (96,256) fp32 -> [o96][k256] bf16 (plain cast; already K contiguous)
__global__ __launch_bounds__(256) void prep_dn_w(const float* __restrict__ src,
                                                 bf16* __restrict__ dst) {
  int idx = blockIdx.x * 256 + threadIdx.x;
  if (idx >= 96 * 256) return;
  dst[idx] = (bf16)src[idx];
}

// ---------------------------------------------------------------------------
// Stem: 7x7 stride-2 conv 32->256, BN+ReLU. Implicit GEMM M=80000 N=256
// K=49x32 (one k-step per tap). Block 128M x 128N (grid.y=2).
// ---------------------------------------------------------------------------
__global__ __launch_bounds__(256) void stem_conv(const bf16* __restrict__ xin,   // [8][206][206][32]
                                                 const bf16* __restrict__ wt,    // [49][256][32]
                                                 const float* __restrict__ g,
                                                 const float* __restrict__ bb,
                                                 const float* __restrict__ mm,
                                                 const float* __restrict__ vv,
                                                 bf16* __restrict__ out) {       // [8][102][102][256]
  __shared__ bf16 Al[2][128][40];
  __shared__ bf16 Bl[2][128][40];

  const int t    = threadIdx.x;
  const int lane = t & 31;
  const int hlf  = lane >> 4;
  const int wid  = t >> 5;
  const int wm   = wid & 3;
  const int wn   = wid >> 2;
  const int mblk = blockIdx.x * 128;
  const int nbas = blockIdx.y * 128;

  // A loader: 2 threads per output pixel, 16 bf16 each
  const int pa = t >> 1;
  const int ha = t & 1;
  const int gp = mblk + pa;
  const int nimg = gp / 10000;
  const int rem  = gp - nimg * 10000;
  const int oh   = rem / 100;
  const int ow   = rem - oh * 100;
  const bf16* p0 = xin + (((size_t)nimg * 206 + oh * 2) * 206 + ow * 2) * 32;

  auto issue = [&](int s, int buf) {
    const int dy = s / 7, dx = s % 7;
    const bf16* ga = p0 + (dy * 206 + dx) * 32 + ha * 16;
    unsigned la = lds_off(&Al[buf][pa][ha * 16]);
    async_b128(la,      ga);
    async_b128(la + 16, ga + 8);
    // B tile: 128 cols x 32 k; 512 b128 slots, 2 per thread
#pragma unroll
    for (int q = 0; q < 2; ++q) {
      const int v   = t + 256 * q;
      const int col = v >> 2;
      const int prt = v & 3;
      async_b128(lds_off(&Bl[buf][col][prt * 8]),
                 wt + ((size_t)s * 256 + nbas + col) * 32 + prt * 8);
    }
  };

  v8f acc[2][4] = {};
  issue(0, 0);

  for (int s = 0; s < 49; ++s) {
    const int cur = s & 1;
    if (s < 48) { issue(s + 1, cur ^ 1); WAIT_ASYNC(4); }
    else        { WAIT_ASYNC(0); }
    __syncthreads();

    v16bf af[2], bf_[4];
#pragma unroll
    for (int i = 0; i < 2; ++i) {
      const int row = wm * 32 + i * 16 + (lane & 15);
      af[i] = make_frag(*(const uint4*)&Al[cur][row][8 * hlf],
                        *(const uint4*)&Al[cur][row][16 + 8 * hlf]);
    }
#pragma unroll
    for (int j = 0; j < 4; ++j) {
      const int col = wn * 64 + j * 16 + (lane & 15);
      bf_[j] = make_frag(*(const uint4*)&Bl[cur][col][8 * hlf],
                         *(const uint4*)&Bl[cur][col][16 + 8 * hlf]);
    }
#pragma unroll
    for (int i = 0; i < 2; ++i)
#pragma unroll
      for (int j = 0; j < 4; ++j)
        acc[i][j] = __builtin_amdgcn_wmma_f32_16x16x32_bf16(
            false, af[i], false, bf_[j], (short)0, acc[i][j], false, false);
    __syncthreads();
  }

  // epilogue: BN + ReLU, padded bf16 NHWC store
#pragma unroll
  for (int j = 0; j < 4; ++j) {
    const int gn = nbas + wn * 64 + j * 16 + (lane & 15);
    const float sc = g[gn] * rsqrtf(vv[gn] + EPS);
    const float sh = bb[gn] - mm[gn] * sc;
#pragma unroll
    for (int i = 0; i < 2; ++i) {
#pragma unroll
      for (int r = 0; r < 8; ++r) {
        const int gm = mblk + wm * 32 + i * 16 + 8 * hlf + r;
        const int n2 = gm / 10000;
        const int r2 = gm - n2 * 10000;
        const int h2 = r2 / 100;
        const int w2 = r2 - h2 * 100;
        float val = fmaxf(acc[i][j][r] * sc + sh, 0.f);
        out[(((size_t)n2 * 102 + h2 + 1) * 102 + w2 + 1) * 256 + gn] = (bf16)val;
      }
    }
  }
}

// ---------------------------------------------------------------------------
// 3x3 conv 256->256 pad 1, BN (+residual) + ReLU. K loop = 9 taps x 8 ksteps.
// Block 128M x 128N (grid.y=2).
// ---------------------------------------------------------------------------
__global__ __launch_bounds__(256) void block_conv(const bf16* __restrict__ xin,  // [8][102][102][256]
                                                  const bf16* __restrict__ wt,   // [9][256][256]
                                                  const float* __restrict__ g,
                                                  const float* __restrict__ bb,
                                                  const float* __restrict__ mm,
                                                  const float* __restrict__ vv,
                                                  const bf16* __restrict__ res,  // may alias out
                                                  bf16* __restrict__ out,
                                                  int use_res) {
  __shared__ bf16 Al[2][128][40];
  __shared__ bf16 Bl[2][128][40];

  const int t    = threadIdx.x;
  const int lane = t & 31;
  const int hlf  = lane >> 4;
  const int wid  = t >> 5;
  const int wm   = wid & 3;
  const int wn   = wid >> 2;
  const int mblk = blockIdx.x * 128;
  const int nbas = blockIdx.y * 128;

  const int pa = t >> 1;
  const int ha = t & 1;
  const int gp = mblk + pa;
  const int nimg = gp / 10000;
  const int rem  = gp - nimg * 10000;
  const int oh   = rem / 100;
  const int ow   = rem - oh * 100;
  const bf16* p0 = xin + (((size_t)nimg * 102 + oh) * 102 + ow) * 256;

  auto issue = [&](int s, int buf) {
    const int tap = s >> 3, ks = s & 7;
    const int dy = tap / 3, dx = tap % 3;
    const bf16* ga = p0 + ((dy * 102 + dx) * 256) + ks * 32 + ha * 16;
    unsigned la = lds_off(&Al[buf][pa][ha * 16]);
    async_b128(la,      ga);
    async_b128(la + 16, ga + 8);
#pragma unroll
    for (int q = 0; q < 2; ++q) {
      const int v   = t + 256 * q;
      const int col = v >> 2;
      const int prt = v & 3;
      async_b128(lds_off(&Bl[buf][col][prt * 8]),
                 wt + ((size_t)tap * 256 + nbas + col) * 256 + ks * 32 + prt * 8);
    }
  };

  v8f acc[2][4] = {};
  issue(0, 0);

  for (int s = 0; s < 72; ++s) {
    const int cur = s & 1;
    if (s < 71) { issue(s + 1, cur ^ 1); WAIT_ASYNC(4); }
    else        { WAIT_ASYNC(0); }
    __syncthreads();

    v16bf af[2], bf_[4];
#pragma unroll
    for (int i = 0; i < 2; ++i) {
      const int row = wm * 32 + i * 16 + (lane & 15);
      af[i] = make_frag(*(const uint4*)&Al[cur][row][8 * hlf],
                        *(const uint4*)&Al[cur][row][16 + 8 * hlf]);
    }
#pragma unroll
    for (int j = 0; j < 4; ++j) {
      const int col = wn * 64 + j * 16 + (lane & 15);
      bf_[j] = make_frag(*(const uint4*)&Bl[cur][col][8 * hlf],
                         *(const uint4*)&Bl[cur][col][16 + 8 * hlf]);
    }
#pragma unroll
    for (int i = 0; i < 2; ++i)
#pragma unroll
      for (int j = 0; j < 4; ++j)
        acc[i][j] = __builtin_amdgcn_wmma_f32_16x16x32_bf16(
            false, af[i], false, bf_[j], (short)0, acc[i][j], false, false);
    __syncthreads();
  }

#pragma unroll
  for (int j = 0; j < 4; ++j) {
    const int gn = nbas + wn * 64 + j * 16 + (lane & 15);
    const float sc = g[gn] * rsqrtf(vv[gn] + EPS);
    const float sh = bb[gn] - mm[gn] * sc;
#pragma unroll
    for (int i = 0; i < 2; ++i) {
#pragma unroll
      for (int r = 0; r < 8; ++r) {
        const int gm = mblk + wm * 32 + i * 16 + 8 * hlf + r;
        const int n2 = gm / 10000;
        const int r2 = gm - n2 * 10000;
        const int h2 = r2 / 100;
        const int w2 = r2 - h2 * 100;
        const size_t ix = (((size_t)n2 * 102 + h2 + 1) * 102 + w2 + 1) * 256 + gn;
        float val = acc[i][j][r] * sc + sh;
        if (use_res) val += (float)res[ix];
        out[ix] = (bf16)fmaxf(val, 0.f);
      }
    }
  }
}

// ---------------------------------------------------------------------------
// Head stage 1: dc = x @ dn_w^T + dn_b (M=80000, N=96, K=256).
// Whole 96x256 bf16 B-panel async-preloaded into LDS once (~50 KB).
// Block tile 128M x 96N; waves 4Mx2N, wave = 2x3 WMMA tiles.
// ---------------------------------------------------------------------------
__global__ __launch_bounds__(256) void dn_gemm(const bf16* __restrict__ xin,  // padded NHWC
                                               const bf16* __restrict__ wt,   // [96][256]
                                               const float* __restrict__ bias,
                                               float* __restrict__ dc) {      // [80000][96]
  __shared__ bf16 Al[2][128][40];
  __shared__ bf16 Bl[96][264];

  const int t    = threadIdx.x;
  const int lane = t & 31;
  const int hlf  = lane >> 4;
  const int wid  = t >> 5;
  const int wm   = wid & 3;
  const int wn   = wid >> 2;
  const int mblk = blockIdx.x * 128;

  const int pa = t >> 1;
  const int ha = t & 1;
  const int gp = mblk + pa;
  const int nimg = gp / 10000;
  const int rem  = gp - nimg * 10000;
  const int oh   = rem / 100;
  const int ow   = rem - oh * 100;
  const bf16* p0 = xin + (((size_t)nimg * 102 + oh + 1) * 102 + ow + 1) * 256;

  // Preload entire B panel: 96*256 bf16 = 3072 x b128, 12 per thread.
#pragma unroll
  for (int q = 0; q < 12; ++q) {
    const int v  = t + 256 * q;       // b128 slot
    const int o  = v >> 5;            // row (col of GEMM)
    const int k0 = (v & 31) * 8;      // k start within row
    async_b128(lds_off(&Bl[o][k0]), wt + (size_t)v * 8);
  }

  auto issueA = [&](int s, int buf) {
    const bf16* ga = p0 + s * 32 + ha * 16;
    unsigned la = lds_off(&Al[buf][pa][ha * 16]);
    async_b128(la,      ga);
    async_b128(la + 16, ga + 8);
  };

  v8f acc[2][3] = {};
  issueA(0, 0);

  for (int s = 0; s < 8; ++s) {
    const int cur = s & 1;
    if (s < 7) { issueA(s + 1, cur ^ 1); WAIT_ASYNC(2); }
    else       { WAIT_ASYNC(0); }
    __syncthreads();

    const int kbase = s * 32;
    v16bf af[2], bf_[3];
#pragma unroll
    for (int i = 0; i < 2; ++i) {
      const int row = wm * 32 + i * 16 + (lane & 15);
      af[i] = make_frag(*(const uint4*)&Al[cur][row][8 * hlf],
                        *(const uint4*)&Al[cur][row][16 + 8 * hlf]);
    }
#pragma unroll
    for (int j = 0; j < 3; ++j) {
      const int col = wn * 48 + j * 16 + (lane & 15);
      bf_[j] = make_frag(*(const uint4*)&Bl[col][kbase + 8 * hlf],
                         *(const uint4*)&Bl[col][kbase + 16 + 8 * hlf]);
    }
#pragma unroll
    for (int i = 0; i < 2; ++i)
#pragma unroll
      for (int j = 0; j < 3; ++j)
        acc[i][j] = __builtin_amdgcn_wmma_f32_16x16x32_bf16(
            false, af[i], false, bf_[j], (short)0, acc[i][j], false, false);
    __syncthreads();
  }

#pragma unroll
  for (int j = 0; j < 3; ++j) {
    const int gn = wn * 48 + j * 16 + (lane & 15);
    const float bs = bias[gn];
#pragma unroll
    for (int i = 0; i < 2; ++i) {
#pragma unroll
      for (int r = 0; r < 8; ++r) {
        const int gm = mblk + wm * 32 + i * 16 + 8 * hlf + r;
        dc[(size_t)gm * 96 + gn] = acc[i][j][r] + bs;
      }
    }
  }
}

// ---------------------------------------------------------------------------
// Head stage 2: per-pixel softmax/zsum + 32->256 GEMM + BN + GELU + scale.
// ---------------------------------------------------------------------------
__device__ __forceinline__ float depth_w(int i) {
  return (1.f + (float)i * (59.f / 63.f)) * (1.f / 30.5f);   // dv / mean(dv)
}

__global__ __launch_bounds__(256) void head_kernel(const float* __restrict__ dc,   // [80000][96]
                                                   const float* __restrict__ cp_w, // [256][32]
                                                   const float* __restrict__ g,
                                                   const float* __restrict__ bb,
                                                   const float* __restrict__ mm,
                                                   const float* __restrict__ vv,
                                                   float* __restrict__ out) {      // [8][256][10000]
  __shared__ float cw[256][33];
  const int t = threadIdx.x;
#pragma unroll
  for (int c = 0; c < 32; ++c) cw[t][c] = cp_w[(size_t)t * 32 + c];
  __syncthreads();

  const int lane = t & 31;
  const int wid  = t >> 5;
  const int p    = blockIdx.x * 8 + wid;

  const float* d = dc + (size_t)p * 96;
  const float l0 = d[lane];
  const float l1 = d[lane + 32];

  float mx = fmaxf(l0, l1);
#pragma unroll
  for (int s = 16; s; s >>= 1) mx = fmaxf(mx, __shfl_xor(mx, s, 32));
  const float e0 = __expf(l0 - mx);
  const float e1 = __expf(l1 - mx);
  float num = e0 * depth_w(lane) + e1 * depth_w(lane + 32);
  float den = e0 + e1;
#pragma unroll
  for (int s = 16; s; s >>= 1) {
    num += __shfl_xor(num, s, 32);
    den += __shfl_xor(den, s, 32);
  }
  const float zs = num / den;

  const float tr = d[64 + lane];
  float av[8] = {0.f, 0.f, 0.f, 0.f, 0.f, 0.f, 0.f, 0.f};
  for (int c = 0; c < 32; ++c) {
    const float tc = __shfl(tr, c, 32);
#pragma unroll
    for (int q = 0; q < 8; ++q) av[q] += tc * cw[q * 32 + lane][c];
  }

  const int nimg = p / 10000;
  const int hw   = p - nimg * 10000;
#pragma unroll
  for (int q = 0; q < 8; ++q) {
    const int o = q * 32 + lane;
    const float sc = g[o] * rsqrtf(vv[o] + EPS);
    const float sh = bb[o] - mm[o] * sc;
    const float x  = av[q] * sc + sh;
    const float ge = 0.5f * x * (1.f + erff(x * 0.70710678118654752f));
    out[((size_t)nimg * 256 + o) * 10000 + hw] = ge * zs;
  }
}

// ---------------------------------------------------------------------------
// Host launcher
// ---------------------------------------------------------------------------
extern "C" void kernel_launch(void* const* d_in, const int* in_sizes, int n_in,
                              void* d_out, int out_size, void* d_ws, size_t ws_size,
                              hipStream_t stream) {
  (void)in_sizes; (void)n_in; (void)out_size; (void)ws_size;

  const float* occ    = (const float*)d_in[0];
  const float* stem_w = (const float*)d_in[1];
  const float* stem_g = (const float*)d_in[2];
  const float* stem_b = (const float*)d_in[3];
  const float* stem_m = (const float*)d_in[4];
  const float* stem_v = (const float*)d_in[5];
  const float* blk_w  = (const float*)d_in[6];
  const float* blk_g  = (const float*)d_in[7];
  const float* blk_b  = (const float*)d_in[8];
  const float* blk_m  = (const float*)d_in[9];
  const float* blk_v  = (const float*)d_in[10];
  const float* dn_w   = (const float*)d_in[11];
  const float* dn_b   = (const float*)d_in[12];
  const float* cp_w   = (const float*)d_in[13];
  const float* cn_g   = (const float*)d_in[14];
  const float* cn_b   = (const float*)d_in[15];
  const float* cn_m   = (const float*)d_in[16];
  const float* cn_v   = (const float*)d_in[17];
  float* out = (float*)d_out;

  // workspace carve-up
  char* ws = (char*)d_ws;
  size_t off = 0;
  auto carve = [&](size_t bytes) -> char* {
    char* p = ws + off;
    off += (bytes + 255) & ~(size_t)255;
    return p;
  };
  const size_t inb_bytes = (size_t)8 * 206 * 206 * 32 * 2;   // padded stem input
  const size_t fea_bytes = (size_t)8 * 102 * 102 * 256 * 2;  // padded features
  bf16*  inb     = (bf16*)carve(inb_bytes);
  bf16*  xb      = (bf16*)carve(fea_bytes);
  bf16*  hb      = (bf16*)carve(fea_bytes);
  bf16*  stem_wt = (bf16*)carve((size_t)49 * 256 * 32 * 2);
  bf16*  blk_wt  = (bf16*)carve((size_t)6 * 9 * 256 * 256 * 2);
  bf16*  dn_wt   = (bf16*)carve((size_t)96 * 256 * 2);
  float* dc      = (float*)carve((size_t)80000 * 96 * 4);

  // --- zero padded buffers (halo must be 0) ---
  {
    int n0 = (int)(inb_bytes / 16);
    int n1 = (int)(fea_bytes / 16);
    zero_buf<<<(n0 + 255) / 256, 256, 0, stream>>>((uint4*)inb, n0);
    zero_buf<<<(n1 + 255) / 256, 256, 0, stream>>>((uint4*)xb, n1);
    zero_buf<<<(n1 + 255) / 256, 256, 0, stream>>>((uint4*)hb, n1);
  }

  // --- prep ---
  prep_input<<<(8 * 32 * 200 * 200 + 255) / 256, 256, 0, stream>>>(occ, inb);
  prep_stem_w<<<(256 * 32 * 49 + 255) / 256, 256, 0, stream>>>(stem_w, stem_wt);
  prep_blk_w<<<(6 * 256 * 256 * 9 + 255) / 256, 256, 0, stream>>>(blk_w, blk_wt);
  prep_dn_w<<<(96 * 256 + 255) / 256, 256, 0, stream>>>(dn_w, dn_wt);

  // --- stem ---
  stem_conv<<<dim3(625, 2), 256, 0, stream>>>(inb, stem_wt, stem_g, stem_b, stem_m, stem_v, xb);

  // --- residual blocks ---
  for (int i = 0; i < 3; ++i) {
    const bf16* w0 = blk_wt + (size_t)(i * 2 + 0) * 9 * 256 * 256;
    const bf16* w1 = blk_wt + (size_t)(i * 2 + 1) * 9 * 256 * 256;
    const int o0 = (i * 2 + 0) * 256;
    const int o1 = (i * 2 + 1) * 256;
    block_conv<<<dim3(625, 2), 256, 0, stream>>>(
        xb, w0, blk_g + o0, blk_b + o0, blk_m + o0, blk_v + o0, (const bf16*)nullptr, hb, 0);
    block_conv<<<dim3(625, 2), 256, 0, stream>>>(
        hb, w1, blk_g + o1, blk_b + o1, blk_m + o1, blk_v + o1, xb, xb, 1);
  }

  // --- head ---
  dn_gemm<<<dim3(625, 1), 256, 0, stream>>>(xb, dn_wt, dn_b, dc);
  head_kernel<<<dim3(10000, 1), 256, 0, stream>>>(dc, cp_w, cn_g, cn_b, cn_m, cn_v, out);
}
